// FeatureEncoder_46385646797578
// MI455X (gfx1250) — matrix operand order repaired and checked by reading, not compiled, not run on hardware
//
#include <hip/hip_runtime.h>

// ---------------------------------------------------------------------------
// Types for CDNA5 WMMA / TDM (gfx1250, wave32)
// ---------------------------------------------------------------------------
typedef __attribute__((ext_vector_type(16))) __bf16 v16bf;
typedef __attribute__((ext_vector_type(8)))  __bf16 v8bf;
typedef __attribute__((ext_vector_type(8)))  float  v8f;
typedef __attribute__((ext_vector_type(4)))  unsigned int u32x4;
typedef __attribute__((ext_vector_type(8)))  int i32x8;
typedef __attribute__((ext_vector_type(4)))  int i32x4;

#if defined(__gfx1250__) && __has_builtin(__builtin_amdgcn_tensor_load_to_lds)
#define USE_TDM 1
#endif

__device__ inline unsigned short f2bf(float f) {
  unsigned u = __float_as_uint(f);
  unsigned r = u + 0x7fffu + ((u >> 16) & 1u);   // round-to-nearest-even
  return (unsigned short)(r >> 16);
}

// ---------------------------------------------------------------------------
// Weight conversion: fp32 OIHW -> bf16 [CoutPad][Kpad], optional spatial flip
// ---------------------------------------------------------------------------
__global__ __launch_bounds__(256)
void wconv_k(const float* __restrict__ w, unsigned short* __restrict__ o,
             int Cout, int CoutPad, int Cin, int Kdim, int Ksz, int Kpad, int flip) {
  int idx = blockIdx.x * 256 + threadIdx.x;
  int tot = CoutPad * Kpad;
  if (idx >= tot) return;
  int co = idx / Kpad, kk = idx - co * Kpad;
  float v = 0.f;
  if (co < Cout && kk < Ksz) {
    int KK = Kdim * Kdim;
    int ci = kk / KK; int r = kk - ci * KK;
    int ky = r / Kdim; int kx = r - ky * Kdim;
    if (flip) { ky = Kdim - 1 - ky; kx = Kdim - 1 - kx; }
    v = w[(((size_t)co * Cin + ci) * Kdim + ky) * Kdim + kx];
  }
  o[idx] = f2bf(v);
}

// ---------------------------------------------------------------------------
// Implicit-GEMM conv via v_wmma_f32_16x16x32_bf16, wave32.
// Block: 128 threads = 4 waves. Tile: 32 Cout x 128 pixels.
// Each wave: 2x2 accumulators -> 4 WMMAs per 32-wide K chunk.
// Weights tile fetched by the Tensor Data Mover (TENSOR_LOAD_TO_LDS),
// issued by wave 0 only, synchronized with s_wait_tensorcnt + barrier.
// ---------------------------------------------------------------------------
#define TM 32
#define TN 128
#define TK 32
#define KMAX 2304   // max Cin*K*K = 256*9

__device__ inline v16bf fragA(const unsigned short* base, int m, int kb) {
  v8bf a0 = *reinterpret_cast<const v8bf*>(base + m * TK + kb);
  v8bf a1 = *reinterpret_cast<const v8bf*>(base + m * TK + kb + 16);
  return __builtin_shufflevector(a0, a1,
      0, 1, 2, 3, 4, 5, 6, 7, 8, 9, 10, 11, 12, 13, 14, 15);
}

__device__ inline v8f wmma_bf16(v16bf a, v16bf b, v8f c) {
  return __builtin_amdgcn_wmma_f32_16x16x32_bf16(
      false, a, false, b, (short)0, c, false, false);
}

#ifdef USE_TDM
// Issue a 2D TDM load: tile [rows=32][cols=32] of 2-byte elems from
// global (row stride = strideElems) into contiguous LDS at lds_off.
__device__ inline void tdm_load_tile(unsigned lds_off, const unsigned short* gptr,
                                     int strideElems, int tensorD0, int tensorD1) {
  unsigned long long ga = (unsigned long long)(uintptr_t)gptr;
  // ---- D# group 0: count=1 | lds_addr | global_addr[56:0] | type=2 --------
  u32x4 g0;
  g0.x = 1u;                                     // count=1, is_restore=0
  g0.y = lds_off;                                // lds_addr
  g0.z = (unsigned)(ga & 0xffffffffull);         // global_addr[31:0]
  g0.w = (unsigned)((ga >> 32) & 0x01ffffffull)  // global_addr[56:32]
         | 0x80000000u;                          // type=2 (bits 127:126)
  // ---- D# group 1 ---------------------------------------------------------
  i32x8 g1;
  g1[0] = 0x00010000;                            // data_size=1 (2 bytes)
  g1[1] = (tensorD0 & 0xffff) << 16;             // tensor_dim0[15:0] @ bits 63:48
  g1[2] = ((unsigned)tensorD0 >> 16)             // tensor_dim0[31:16]
        | ((tensorD1 & 0xffff) << 16);           // tensor_dim1[15:0]
  g1[3] = ((unsigned)tensorD1 >> 16)             // tensor_dim1[31:16]
        | (TK << 16);                            // tile_dim0 = 32
  g1[4] = TM;                                    // tile_dim1 = 32 (tile_dim2 = 0)
  g1[5] = strideElems;                           // tensor_dim0_stride[31:0]
  g1[6] = 0;                                     // stride[47:32]=0, dim1_stride lo
  g1[7] = 0;
  i32x4 z4 = {};
#if __clang_major__ >= 23
  i32x8 z8 = {};
  __builtin_amdgcn_tensor_load_to_lds(g0, g1, z4, z4, z8, 0);
#else
  __builtin_amdgcn_tensor_load_to_lds(g0, g1, z4, z4, 0);
#endif
}
#endif

__global__ __launch_bounds__(128)
void conv_wmma(const float* __restrict__ x, const unsigned short* __restrict__ wbf,
               const float* __restrict__ bias, float* __restrict__ y,
               int Cin, int Cout, int H, int W, int OH, int OW,
               int Kdim, int stride, int pad, int mode, int Ksz, int Kpad) {
  __shared__ alignas(32) unsigned short lA[TM * TK];   // [m][k], 2 KB
  __shared__ alignas(32) unsigned short lB[TN * TK];   // [n][k], 8 KB
  __shared__ int dec[KMAX];                            // k -> (ci,ky,kx)

  const int tid  = threadIdx.x;
  const int lane = tid & 31;
  const int wave = tid >> 5;
  const int b    = blockIdx.z;
  const int co0  = blockIdx.y * TM;
  const int pix0 = blockIdx.x * TN;
  const int npix = OH * OW;
  const int KK   = Kdim * Kdim;

  // ---- one-time K decode table --------------------------------------------
  for (int k = tid; k < Kpad; k += 128) {
    int ci = k / KK; int r = k - ci * KK;
    int ky = r / Kdim; int kx = r - ky * Kdim;
    dec[k] = (ci << 8) | (ky << 4) | kx;
  }

  // ---- this thread owns one im2col row (pixel) -----------------------------
  const int   nB  = pix0 + tid;
  const bool  pvB = nB < npix;
  const int   oyB = nB / OW;
  const int   oxB = nB - oyB * OW;
  const float* xb = x + (size_t)b * Cin * H * W;

#ifdef USE_TDM
  const unsigned ldsAoff = (unsigned)(uintptr_t)(void*)&lA[0];
#endif

  __syncthreads();

  v8f acc00 = {}, acc01 = {}, acc10 = {}, acc11 = {};

  for (int k0 = 0; k0 < Kpad; k0 += TK) {
    // ---- stage A tile (weights) via the Tensor Data Mover -------------------
#ifdef USE_TDM
    if (wave == 0) {
      tdm_load_tile(ldsAoff, wbf + (size_t)co0 * Kpad + k0, Kpad, Kpad, TM);
      __builtin_amdgcn_s_wait_tensorcnt(0);
    }
#else
    {
      int m  = tid >> 2;
      int kk = (tid & 3) * 8;
      const unsigned short* src = wbf + (size_t)(co0 + m) * Kpad + k0 + kk;
#pragma unroll
      for (int j = 0; j < 8; ++j) lA[m * TK + kk + j] = src[j];
    }
#endif
    // ---- stage B tile: one pixel row (32 k) per thread ----------------------
    {
#pragma unroll 4
      for (int kk = 0; kk < TK; ++kk) {
        int kg = k0 + kk;
        float v = 0.f;
        if (pvB && kg < Ksz) {
          int d  = dec[kg];
          int ci = d >> 8, ky = (d >> 4) & 15, kx = d & 15;
          int iy, ix; bool ok = true;
          if (mode == 0) {                 // zero pad, stride s
            iy = oyB * stride + ky - pad;
            ix = oxB * stride + kx - pad;
            ok = (iy >= 0) & (iy < H) & (ix >= 0) & (ix < W);
          } else if (mode == 1) {          // reflect pad, stride 1
            iy = oyB + ky - pad;
            ix = oxB + kx - pad;
            iy = iy < 0 ? -iy : (iy >= H ? 2 * H - 2 - iy : iy);
            ix = ix < 0 ? -ix : (ix >= W ? 2 * W - 2 - ix : ix);
          } else {                         // transposed: lhs_dilation=2, pad(1,2)
            int ty = oyB + ky - 1, tx = oxB + kx - 1;
            ok = (ty >= 0) & (tx >= 0) & ((ty & 1) == 0) & ((tx & 1) == 0);
            iy = ty >> 1; ix = tx >> 1;
            ok = ok & (iy < H) & (ix < W);
          }
          if (ok) v = xb[((size_t)ci * H + iy) * W + ix];
        }
        lB[tid * TK + kk] = f2bf(v);
      }
    }
    __syncthreads();

    // ---- fragments + 4 WMMAs per wave --------------------------------------
    {
      int mlo = lane & 15;
      int kb  = (lane >> 4) * 8;
      int kh  = (lane >> 4) * 16;
      v16bf a0 = fragA(lA, mlo, kb);
      v16bf a1 = fragA(lA + 16 * TK, mlo, kb);

      int pn0 = wave * 32 + (lane & 15);
      v16bf b0 = fragA(lB, pn0, kb);
      v16bf b1 = fragA(lB, pn0 + 16, kb);

      acc00 = wmma_bf16(a0, b0, acc00);
      acc01 = wmma_bf16(a0, b1, acc01);
      acc10 = wmma_bf16(a1, b0, acc10);
      acc11 = wmma_bf16(a1, b1, acc11);
    }
    __syncthreads();
  }

  // ---- store 4 D tiles ----------------------------------------------------
  {
    const int hi = lane >> 4;
    const int nn = lane & 15;
    const int nbase = pix0 + wave * 32;
    const size_t ybase = (size_t)b * Cout * npix;
    int n0 = nbase + nn, n1 = nbase + 16 + nn;
#pragma unroll
    for (int v = 0; v < 8; ++v) {
      int coA = co0 + v + 8 * hi;
      int coB = coA + 16;
      if (coA < Cout) {
        float bA = bias[coA];
        if (n0 < npix) y[ybase + (size_t)coA * npix + n0] = acc00[v] + bA;
        if (n1 < npix) y[ybase + (size_t)coA * npix + n1] = acc01[v] + bA;
      }
      if (coB < Cout) {
        float bB = bias[coB];
        if (n0 < npix) y[ybase + (size_t)coB * npix + n0] = acc10[v] + bB;
        if (n1 < npix) y[ybase + (size_t)coB * npix + n1] = acc11[v] + bB;
      }
    }
  }
}

// ---------------------------------------------------------------------------
// Instance norm + ReLU, split three ways for occupancy:
//   in_part_k : (S, B*C) blocks -> per-split sum/sumsq partials
//   in_final_k: (B*C) blocks    -> mu, rstd
//   in_apply_k: elementwise normalize + relu
// ---------------------------------------------------------------------------
#define INSPLIT 64

__global__ __launch_bounds__(256)
void in_part_k(const float* __restrict__ x, float* __restrict__ part, int HW) {
  __shared__ float s1[256], s2[256];
  int bc = blockIdx.y, s = blockIdx.x, tid = threadIdx.x;
  const float* p = x + (size_t)bc * HW;
  int chunk = (HW + INSPLIT - 1) / INSPLIT;
  int i0 = s * chunk;
  int i1 = i0 + chunk; if (i1 > HW) i1 = HW;
  float sum = 0.f, ss = 0.f;
  for (int i = i0 + tid; i < i1; i += 256) { float v = p[i]; sum += v; ss += v * v; }
  s1[tid] = sum; s2[tid] = ss;
  __syncthreads();
  for (int o = 128; o > 0; o >>= 1) {
    if (tid < o) { s1[tid] += s1[tid + o]; s2[tid] += s2[tid + o]; }
    __syncthreads();
  }
  if (tid == 0) {
    part[((size_t)bc * INSPLIT + s) * 2 + 0] = s1[0];
    part[((size_t)bc * INSPLIT + s) * 2 + 1] = s2[0];
  }
}

__global__ __launch_bounds__(64)
void in_final_k(const float* __restrict__ part, float* __restrict__ stats, int HW) {
  __shared__ float s1[64], s2[64];
  int bc = blockIdx.x, tid = threadIdx.x;
  s1[tid] = part[((size_t)bc * INSPLIT + tid) * 2 + 0];
  s2[tid] = part[((size_t)bc * INSPLIT + tid) * 2 + 1];
  __syncthreads();
  for (int o = 32; o > 0; o >>= 1) {
    if (tid < o) { s1[tid] += s1[tid + o]; s2[tid] += s2[tid + o]; }
    __syncthreads();
  }
  if (tid == 0) {
    float m = s1[0] / (float)HW;
    float var = s2[0] / (float)HW - m * m;
    stats[bc * 2 + 0] = m;
    stats[bc * 2 + 1] = rsqrtf(var + 1e-5f);
  }
}

__global__ __launch_bounds__(256)
void in_apply_k(float* __restrict__ x, const float* __restrict__ stats, int HW) {
  int bc = blockIdx.y;
  int i = blockIdx.x * 256 + threadIdx.x;
  if (i >= HW) return;
  float mu = stats[bc * 2 + 0], rs = stats[bc * 2 + 1];
  float v = (x[(size_t)bc * HW + i] - mu) * rs;
  x[(size_t)bc * HW + i] = v > 0.f ? v : 0.f;
}

__global__ __launch_bounds__(256)
void tanh_k(float* __restrict__ x, int n) {
  int i = blockIdx.x * 256 + threadIdx.x;
  if (i < n) x[i] = tanhf(x[i]);
}

// ---------------------------------------------------------------------------
// Segment-mean pooling over instance map (NUM_INST=32, 3 channels, B=2)
// ---------------------------------------------------------------------------
#define NUM_INST 32

__global__ void pool_zero(float* sums, float* cnts) {
  int t = threadIdx.x;
  if (t < 2 * NUM_INST * 3) sums[t] = 0.f;
  if (t < 2 * NUM_INST)     cnts[t] = 0.f;
}

__global__ __launch_bounds__(256)
void pool_acc(const float* __restrict__ x, const int* __restrict__ inst,
              float* __restrict__ sums, float* __restrict__ cnts, int HW) {
  __shared__ float ls[NUM_INST * 3];
  __shared__ float lc[NUM_INST];
  int b = blockIdx.y, tid = threadIdx.x;
  for (int i = tid; i < NUM_INST * 3; i += 256) ls[i] = 0.f;
  for (int i = tid; i < NUM_INST; i += 256)     lc[i] = 0.f;
  __syncthreads();
  int stridegrid = gridDim.x * 256;
  for (int i = blockIdx.x * 256 + tid; i < HW; i += stridegrid) {
    int id = inst[(size_t)b * HW + i];
    atomicAdd(&lc[id], 1.f);
    for (int c = 0; c < 3; ++c)
      atomicAdd(&ls[id * 3 + c], x[((size_t)b * 3 + c) * HW + i]);
  }
  __syncthreads();
  for (int i = tid; i < NUM_INST * 3; i += 256) atomicAdd(&sums[b * NUM_INST * 3 + i], ls[i]);
  for (int i = tid; i < NUM_INST; i += 256)     atomicAdd(&cnts[b * NUM_INST + i], lc[i]);
}

__global__ __launch_bounds__(256)
void pool_out(const int* __restrict__ inst, const float* __restrict__ sums,
              const float* __restrict__ cnts, float* __restrict__ out, int HW) {
  int b = blockIdx.y;
  int i = blockIdx.x * 256 + threadIdx.x;
  if (i >= HW) return;
  int id = inst[(size_t)b * HW + i];
  float c = cnts[b * NUM_INST + id]; c = c < 1.f ? 1.f : c;
  for (int ch = 0; ch < 3; ++ch)
    out[((size_t)b * 3 + ch) * HW + i] = sums[(b * NUM_INST + id) * 3 + ch] / c;
}

// ---------------------------------------------------------------------------
// Orchestration
// ---------------------------------------------------------------------------
extern "C" void kernel_launch(void* const* d_in, const int* in_sizes, int n_in,
                              void* d_out, int out_size, void* d_ws, size_t ws_size,
                              hipStream_t stream) {
  (void)in_sizes; (void)n_in; (void)out_size; (void)ws_size;

  const float* input = (const float*)d_in[0];
  const int*   inst  = (const int*)d_in[1];
  const float* Wf[10]; const float* Bf[10];
  for (int i = 0; i < 10; ++i) {
    Wf[i] = (const float*)d_in[2 + 2 * i];
    Bf[i] = (const float*)d_in[3 + 2 * i];
  }

  char* ws = (char*)d_ws;
  float* buf0 = (float*)ws;                            // 64 MB (max act)
  float* buf1 = (float*)(ws + 67108864);               // 33.5 MB
  unsigned short* wbase = (unsigned short*)(ws + 100663296);
  float* psum  = (float*)(ws + 102760448);             // 2*32*3
  float* pcnt  = psum + 2 * NUM_INST * 3;              // 2*32
  float* part  = (float*)(ws + 102764544);             // 512*64*2 floats
  float* stats = (float*)(ws + 103026688);             // 512*2 floats

  struct L { int Cin, Cout, K, stride, pad, mode, H, W, OH, OW; };
  const L ls[10] = {
    {  3,  16, 7, 1, 3, 1, 512, 1024, 512, 1024},   // e0 reflect 7x7
    { 16,  32, 3, 2, 1, 0, 512, 1024, 256,  512},   // e1
    { 32,  64, 3, 2, 1, 0, 256,  512, 128,  256},   // e2
    { 64, 128, 3, 2, 1, 0, 128,  256,  64,  128},   // e3
    {128, 256, 3, 2, 1, 0,  64,  128,  32,   64},   // e4
    {256, 128, 3, 1, 1, 2,  32,   64,  64,  128},   // d0 transposed
    {128,  64, 3, 1, 1, 2,  64,  128, 128,  256},   // d1
    { 64,  32, 3, 1, 1, 2, 128,  256, 256,  512},   // d2
    { 32,  16, 3, 1, 1, 2, 256,  512, 512, 1024},   // d3
    { 16,   3, 7, 1, 3, 1, 512, 1024, 512, 1024},   // d4 reflect 7x7
  };

  // ---- convert weights to padded bf16 (flip for transposed layers) --------
  unsigned short* wptr[10];
  int KszA[10], KpadA[10], CoutPadA[10];
  size_t off = 0;
  for (int i = 0; i < 10; ++i) {
    KszA[i]     = ls[i].Cin * ls[i].K * ls[i].K;
    KpadA[i]    = (KszA[i] + 31) & ~31;
    CoutPadA[i] = (ls[i].Cout + 31) & ~31;
    wptr[i] = wbase + off;
    off += (size_t)CoutPadA[i] * KpadA[i];
    int tot = CoutPadA[i] * KpadA[i];
    wconv_k<<<(tot + 255) / 256, 256, 0, stream>>>(
        Wf[i], wptr[i], ls[i].Cout, CoutPadA[i], ls[i].Cin,
        ls[i].K, KszA[i], KpadA[i], ls[i].mode == 2 ? 1 : 0);
  }

  // ---- conv stack ---------------------------------------------------------
  const float* cur = input;
  for (int i = 0; i < 10; ++i) {
    float* dst = (i & 1) ? buf1 : buf0;
    int npix = ls[i].OH * ls[i].OW;
    dim3 grid((npix + TN - 1) / TN, CoutPadA[i] / TM, 2);
    conv_wmma<<<grid, 128, 0, stream>>>(
        cur, wptr[i], Bf[i], dst,
        ls[i].Cin, ls[i].Cout, ls[i].H, ls[i].W, ls[i].OH, ls[i].OW,
        ls[i].K, ls[i].stride, ls[i].pad, ls[i].mode, KszA[i], KpadA[i]);
    if (i < 9) {
      int BC = 2 * ls[i].Cout;
      in_part_k <<<dim3(INSPLIT, BC), 256, 0, stream>>>(dst, part, npix);
      in_final_k<<<BC, 64, 0, stream>>>(part, stats, npix);
      in_apply_k<<<dim3((npix + 255) / 256, BC), 256, 0, stream>>>(dst, stats, npix);
    } else {
      int n = 2 * 3 * npix;
      tanh_k<<<(n + 255) / 256, 256, 0, stream>>>(dst, n);
    }
    cur = dst;
  }

  // ---- instance-mean pooling ---------------------------------------------
  const int HW = 512 * 1024;
  pool_zero<<<1, 256, 0, stream>>>(psum, pcnt);
  pool_acc<<<dim3(256, 2), 256, 0, stream>>>(cur, inst, psum, pcnt, HW);
  pool_out<<<dim3((HW + 255) / 256, 2), 256, 0, stream>>>(inst, psum, pcnt, (float*)d_out, HW);
}
